// PGTJAttention_26568667693705
// MI455X (gfx1250) — matrix operand order repaired and checked by reading, not compiled, not run on hardware
//
#include <hip/hip_runtime.h>
#include <hip/hip_bf16.h>
#include <math.h>

// ---------------------------------------------------------------------------
// CDNA5 feature probes (device pass defines __gfx1250__; host pass falls back)
// ---------------------------------------------------------------------------
#if defined(__gfx1250__) && __has_builtin(__builtin_amdgcn_global_load_async_to_lds_b128) && __has_builtin(__builtin_amdgcn_s_wait_asynccnt)
#define USE_ASYNC_LDS 1
#else
#define USE_ASYNC_LDS 0
#endif

// ---------------------------------------------------------------------------
// Types for CDNA5 WMMA fragments (wave32).
// ---------------------------------------------------------------------------
typedef __attribute__((ext_vector_type(16))) _Float16 v16h;
typedef __attribute__((ext_vector_type(8)))  _Float16 v8h;
typedef __attribute__((ext_vector_type(4)))  _Float16 v4h;
typedef __attribute__((ext_vector_type(8)))  float    v8f;
typedef __attribute__((ext_vector_type(4)))  float    v4f;
typedef __attribute__((ext_vector_type(4)))  int      v4i;

#if USE_ASYNC_LDS
#define AS1 __attribute__((address_space(1)))
#define AS3 __attribute__((address_space(3)))
// Async memory -> LDS copy of 16 bytes per lane (GLOBAL_LOAD_ASYNC_TO_LDS_B128,
// tracked by ASYNCcnt; no VGPR data round-trip).
// Builtin signature (from clang): (v4i AS1*, v4i AS3*, imm offset, imm cpol).
__device__ __forceinline__ void async_cp16(const _Float16* g, _Float16* l) {
    __builtin_amdgcn_global_load_async_to_lds_b128(
        (AS1 v4i*)(void*)(g), (AS3 v4i*)(void*)(l), /*offset=*/0, /*cpol=*/0);
}
__device__ __forceinline__ void wait_async0() {
    __builtin_amdgcn_s_wait_asynccnt(0);
}
#endif

// Concatenate two 16-byte chunks into one 32-byte fragment register group.
__device__ __forceinline__ v16h frag_cat(const _Float16* p0, const _Float16* p1) {
    v8h a = *(const v8h*)p0;
    v8h b = *(const v8h*)p1;
    return __builtin_shufflevector(a, b, 0,1,2,3,4,5,6,7,8,9,10,11,12,13,14,15);
}

__device__ __forceinline__ v8f wmma_f16(v16h a, v16h b, v8f c) {
    // D = A(16x32 f16) * B(32x16 f16) + C(16x16 f32)
    return __builtin_amdgcn_wmma_f32_16x16x32_f16(
        /*neg_a=*/false, a, /*neg_b=*/false, b,
        /*c_mod=*/(short)0, c, /*reuse_a=*/false, /*reuse_b=*/false);
}

// ---------------------------------------------------------------------------
// Problem constants (match the reference).
// ---------------------------------------------------------------------------
#define BB   2
#define SS   2048
#define DD   1024
#define HH   16
#define HDIM 64
#define ROT  32
#define MM   (BB * SS)      // 4096 rows

// ---------------------------------------------------------------------------
// Kernel 1: fp32 -> fp16 conversion (vector x4).
// ---------------------------------------------------------------------------
__global__ void cvt_f32_f16(const float* __restrict__ in,
                            _Float16* __restrict__ out, int n4) {
    int i = blockIdx.x * blockDim.x + threadIdx.x;
    if (i < n4) {
        v4f v = ((const v4f*)in)[i];
        ((v4h*)out)[i] = __builtin_convertvector(v, v4h);
    }
}

// ---------------------------------------------------------------------------
// Kernel 2: tiled WMMA GEMM  C(MxN,f32) = A(MxK,f16) * W(NxK,f16)^T
// Block = 256 threads = 8 waves; tile 128x64, BK = 32, double-buffered LDS,
// async (memory -> LDS) staging when the toolchain exposes the builtin.
// Wave (wm,wn) in 4x2 grid owns a 32x32 sub-tile = 2x2 WMMA fragments.
// ---------------------------------------------------------------------------
__global__ __launch_bounds__(256)
void gemm_awt_wmma(const _Float16* __restrict__ A,
                   const _Float16* __restrict__ W,
                   float* __restrict__ C, int M, int N, int K) {
    __shared__ _Float16 As[2][128][32];  // 2 x 8 KB
    __shared__ _Float16 Bs[2][64][32];   // 2 x 4 KB

    const int tid  = threadIdx.x;
    const int lane = tid & 31;
    const int wave = tid >> 5;
    const int wm   = wave >> 1;          // 0..3
    const int wn   = wave & 1;           // 0..1
    const int m0   = blockIdx.y * 128;
    const int n0   = blockIdx.x * 64;
    const int lm   = lane & 15;
    const int lh   = lane >> 4;

    v8f acc[2][2] = {};

    const int srow = tid >> 2;           // 0..63
    const int sseg = (tid & 3) * 8;      // 0,8,16,24

    const _Float16* gA0 = A + (size_t)(m0 + srow) * K + sseg;
    const _Float16* gA1 = A + (size_t)(m0 + srow + 64) * K + sseg;
    const _Float16* gB  = W + (size_t)(n0 + srow) * K + sseg;

    auto stage = [&](int buf, int k0) {
#if USE_ASYNC_LDS
        async_cp16(gA0 + k0, &As[buf][srow][sseg]);
        async_cp16(gA1 + k0, &As[buf][srow + 64][sseg]);
        async_cp16(gB  + k0, &Bs[buf][srow][sseg]);
#else
        *(v8h*)&As[buf][srow][sseg]      = *(const v8h*)(gA0 + k0);
        *(v8h*)&As[buf][srow + 64][sseg] = *(const v8h*)(gA1 + k0);
        *(v8h*)&Bs[buf][srow][sseg]      = *(const v8h*)(gB + k0);
#endif
    };

    // Prologue: stage tile 0.
    stage(0, 0);
#if USE_ASYNC_LDS
    wait_async0();
#endif
    __syncthreads();

    int cur = 0;
    for (int k0 = 0; k0 < K; k0 += 32) {
        // Kick off the next tile's staging while we compute on the current one.
        if (k0 + 32 < K) stage(cur ^ 1, k0 + 32);

        // A fragment: lane(m=lm, h=lh) needs K = h*8..h*8+7 and 16+h*8..+7.
        v16h af0 = frag_cat(&As[cur][wm * 32 + lm][lh * 8],
                            &As[cur][wm * 32 + lm][lh * 8 + 16]);
        v16h af1 = frag_cat(&As[cur][wm * 32 + 16 + lm][lh * 8],
                            &As[cur][wm * 32 + 16 + lm][lh * 8 + 16]);
        // B fragment: lane(n=lm, h=lh) needs K = h*8..h*8+7 and 16+h*8..+7.
        v16h bf0 = frag_cat(&Bs[cur][wn * 32 + lm][lh * 8],
                            &Bs[cur][wn * 32 + lm][lh * 8 + 16]);
        v16h bf1 = frag_cat(&Bs[cur][wn * 32 + 16 + lm][lh * 8],
                            &Bs[cur][wn * 32 + 16 + lm][lh * 8 + 16]);

        acc[0][0] = wmma_f16(af0, bf0, acc[0][0]);
        acc[0][1] = wmma_f16(af0, bf1, acc[0][1]);
        acc[1][0] = wmma_f16(af1, bf0, acc[1][0]);
        acc[1][1] = wmma_f16(af1, bf1, acc[1][1]);

#if USE_ASYNC_LDS
        wait_async0();                 // this wave's async stores have landed
#endif
        __syncthreads();               // everyone's staging + reads complete
        cur ^= 1;
    }

    // D fragment layout: VGPR r -> row (r + 8*lh), col = lm.
    #pragma unroll
    for (int i = 0; i < 2; ++i)
        #pragma unroll
        for (int j = 0; j < 2; ++j) {
            const int mb = m0 + wm * 32 + i * 16 + 8 * lh;
            const int nb = n0 + wn * 32 + j * 16 + lm;
            #pragma unroll
            for (int r = 0; r < 8; ++r)
                C[(size_t)(mb + r) * N + nb] = acc[i][j][r];
        }
}

// ---------------------------------------------------------------------------
// Kernel 3: RoPE + repack to attention-friendly f16 layouts.
//   q,k,v: (B, S, H*HD) fp32 from the projection GEMMs.
//   qp,kp: (B, H, S, HD) f16   (q pre-scaled by 1/sqrt(HD))
//   vp:    (B, H, HD, S) f16   (key-contiguous for PV B-fragments)
// ---------------------------------------------------------------------------
__global__ void rope_pack(const float* __restrict__ q,
                          const float* __restrict__ k,
                          const float* __restrict__ v,
                          _Float16* __restrict__ qp,
                          _Float16* __restrict__ kp,
                          _Float16* __restrict__ vp) {
    const int idx = blockIdx.x * blockDim.x + threadIdx.x;  // B*S*H*HD threads
    const int d = idx & (HDIM - 1);
    const int h = (idx >> 6) & (HH - 1);
    const int s = (idx >> 10) & (SS - 1);
    const int b = idx >> 21;

    const size_t src = ((size_t)(b * SS + s)) * DD + h * HDIM + d;
    float qv = q[src];
    float kv = k[src];

    if (d < ROT) {
        const int   pi  = d >> 1;
        const float inv = __powf(10000.0f, -(float)(2 * pi) / (float)ROT);
        const float ang = (float)s * inv;
        const float sn  = __sinf(ang);
        const float cs  = __cosf(ang);
        const float qo  = q[src ^ 1];   // pair partner (d is even<->odd)
        const float ko  = k[src ^ 1];
        if (d & 1) { qv = qv * cs + qo * sn;  kv = kv * cs + ko * sn; }
        else       { qv = qv * cs - qo * sn;  kv = kv * cs - ko * sn; }
    }

    const size_t dq = (((size_t)(b * HH + h) * SS + s) * HDIM) + d;
    qp[dq] = (_Float16)(qv * 0.125f);   // fold 1/sqrt(64) into Q
    kp[dq] = (_Float16)kv;
    const size_t dv = (((size_t)(b * HH + h) * HDIM + d) * SS) + s;
    vp[dv] = (_Float16)v[src];
}

// ---------------------------------------------------------------------------
// Kernel 4: causal flash attention, WMMA QK^T and PV, online softmax.
// Block = 128 threads = 4 independent waves; each wave owns 16 query rows.
// ---------------------------------------------------------------------------
__global__ __launch_bounds__(128)
void flash_attn_wmma(const _Float16* __restrict__ qp,
                     const _Float16* __restrict__ kp,
                     const _Float16* __restrict__ vp,
                     _Float16* __restrict__ attn) {
    const int b    = blockIdx.z;
    const int h    = blockIdx.y;
    const int wave = threadIdx.x >> 5;
    const int lane = threadIdx.x & 31;
    const int lm   = lane & 15;
    const int lh   = lane >> 4;
    const int q0   = blockIdx.x * 64 + wave * 16;

    const _Float16* Q  = qp + ((size_t)(b * HH + h) * SS) * HDIM;
    const _Float16* Kt = kp + ((size_t)(b * HH + h) * SS) * HDIM;
    const _Float16* V  = vp + ((size_t)(b * HH + h) * HDIM) * SS;

    __shared__ _Float16 Plds[4][16][32];          // per-wave P tile, no barriers
    _Float16 (*P)[32] = Plds[wave];

    // Q fragments (fixed for the whole key loop): 2 k-steps over HD=64.
    v16h qa[2];
    #pragma unroll
    for (int kk = 0; kk < 2; ++kk) {
        const _Float16* p0 = Q + (size_t)(q0 + lm) * HDIM + kk * 32 + lh * 8;
        qa[kk] = frag_cat(p0, p0 + 16);
    }

    v8f acc[4] = {};
    float rmax[8], rsum[8];
    #pragma unroll
    for (int r = 0; r < 8; ++r) { rmax[r] = -INFINITY; rsum[r] = 0.0f; }

    const int kend = q0 + 16;                     // causal bound for this wave
    for (int t0 = 0; t0 < kend; t0 += 32) {
        // Prefetch next key/value tiles (global_prefetch_b8 on gfx1250).
        if (t0 + 32 < kend) {
            __builtin_prefetch(Kt + (size_t)(t0 + 32 + lm) * HDIM + lh * 16, 0, 3);
            __builtin_prefetch(V + (size_t)(lane * 2) * SS + t0 + 32, 0, 3);
        }
        // ---- scores S(16x32) = Q(16x64) * K^T --------------------------------
        v8f s[2] = {};
        #pragma unroll
        for (int nc = 0; nc < 2; ++nc) {
            const _Float16* kb = Kt + (size_t)(t0 + nc * 16 + lm) * HDIM + lh * 8;
            #pragma unroll
            for (int kk = 0; kk < 2; ++kk) {
                v16h bf = frag_cat(kb + kk * 32, kb + kk * 32 + 16);
                s[nc] = wmma_f16(qa[kk], bf, s[nc]);
            }
        }
        // ---- causal mask (only tiles straddling the diagonal) ---------------
        if (t0 + 31 > q0) {
            #pragma unroll
            for (int nc = 0; nc < 2; ++nc) {
                const int key = t0 + nc * 16 + lm;
                #pragma unroll
                for (int r = 0; r < 8; ++r)
                    if (key > q0 + r + 8 * lh) s[nc][r] = -INFINITY;
            }
        }
        // ---- online softmax (row = r + 8*lh, spread over 16 lanes) ----------
        #pragma unroll
        for (int r = 0; r < 8; ++r) {
            float vmax = fmaxf(s[0][r], s[1][r]);
            #pragma unroll
            for (int off = 8; off; off >>= 1)
                vmax = fmaxf(vmax, __shfl_xor(vmax, off, 16));
            const float nm    = fmaxf(rmax[r], vmax);
            const float alpha = __expf(rmax[r] - nm);
            rmax[r] = nm;
            const float p0v = __expf(s[0][r] - nm);
            const float p1v = __expf(s[1][r] - nm);
            float psum = p0v + p1v;
            #pragma unroll
            for (int off = 8; off; off >>= 1)
                psum += __shfl_xor(psum, off, 16);
            rsum[r] = rsum[r] * alpha + psum;
            #pragma unroll
            for (int ct = 0; ct < 4; ++ct) acc[ct][r] *= alpha;
            // D-layout -> LDS so we can re-read as an A-fragment.
            P[r + 8 * lh][lm]      = (_Float16)p0v;
            P[r + 8 * lh][16 + lm] = (_Float16)p1v;
        }
        // ---- PV: acc(16x64) += P(16x32) * V(32x64) --------------------------
        v16h pf = frag_cat(&P[lm][lh * 8], &P[lm][lh * 8 + 16]);
        #pragma unroll
        for (int ct = 0; ct < 4; ++ct) {
            const _Float16* vb = V + (size_t)(ct * 16 + lm) * SS + t0 + lh * 8;
            v16h bf = frag_cat(vb, vb + 16);
            acc[ct] = wmma_f16(pf, bf, acc[ct]);
        }
    }

    // ---- epilogue: normalize, write (B, S, H*HD) f16 for the final GEMM ----
    #pragma unroll
    for (int ct = 0; ct < 4; ++ct)
        #pragma unroll
        for (int r = 0; r < 8; ++r) {
            const int m = q0 + r + 8 * lh;
            const size_t o = ((size_t)(b * SS + m)) * DD + h * HDIM + ct * 16 + lm;
            attn[o] = (_Float16)(acc[ct][r] / rsum[r]);
        }
}

// ---------------------------------------------------------------------------
// Host-side orchestration.
// ---------------------------------------------------------------------------
extern "C" void kernel_launch(void* const* d_in, const int* in_sizes, int n_in,
                              void* d_out, int out_size, void* d_ws, size_t ws_size,
                              hipStream_t stream) {
    const float* hs = (const float*)d_in[0];
    const float* wq = (const float*)d_in[1];
    const float* wk = (const float*)d_in[2];
    const float* wv = (const float*)d_in[3];
    const float* wo = (const float*)d_in[4];
    float* out = (float*)d_out;

    const size_t NMK = (size_t)MM * DD;    // 4096*1024 elements
    const size_t NDD = (size_t)DD * DD;    // 1024*1024 elements

    char* ws = (char*)d_ws;
    size_t off = 0;
    auto take = [&](size_t bytes) -> char* {
        char* p = ws + off;
        off += (bytes + 255) & ~(size_t)255;
        return p;
    };

    _Float16* hs16 = (_Float16*)take(NMK * 2);
    _Float16* wq16 = (_Float16*)take(NDD * 2);
    _Float16* wk16 = (_Float16*)take(NDD * 2);
    _Float16* wv16 = (_Float16*)take(NDD * 2);
    _Float16* wo16 = (_Float16*)take(NDD * 2);
    float*    qf   = (float*)take(NMK * 4);
    float*    kf   = (float*)take(NMK * 4);
    float*    vf   = (float*)take(NMK * 4);
    _Float16* qpk  = (_Float16*)take(NMK * 2);
    _Float16* kpk  = (_Float16*)take(NMK * 2);
    _Float16* vpk  = (_Float16*)take(NMK * 2);
    // attn reuses the (now dead) qf region after rope_pack runs.
    _Float16* at16 = (_Float16*)qf;

    // 1) fp32 -> fp16 conversions
    {
        const int t = 256;
        cvt_f32_f16<<<(int)(NMK / 4 + t - 1) / t, t, 0, stream>>>(hs, hs16, (int)(NMK / 4));
        cvt_f32_f16<<<(int)(NDD / 4 + t - 1) / t, t, 0, stream>>>(wq, wq16, (int)(NDD / 4));
        cvt_f32_f16<<<(int)(NDD / 4 + t - 1) / t, t, 0, stream>>>(wk, wk16, (int)(NDD / 4));
        cvt_f32_f16<<<(int)(NDD / 4 + t - 1) / t, t, 0, stream>>>(wv, wv16, (int)(NDD / 4));
        cvt_f32_f16<<<(int)(NDD / 4 + t - 1) / t, t, 0, stream>>>(wo, wo16, (int)(NDD / 4));
    }

    // 2) Q/K/V projections: C = hs @ W^T
    {
        dim3 grid(DD / 64, MM / 128);
        gemm_awt_wmma<<<grid, 256, 0, stream>>>(hs16, wq16, qf, MM, DD, DD);
        gemm_awt_wmma<<<grid, 256, 0, stream>>>(hs16, wk16, kf, MM, DD, DD);
        gemm_awt_wmma<<<grid, 256, 0, stream>>>(hs16, wv16, vf, MM, DD, DD);
    }

    // 3) RoPE + repack
    {
        const int total = BB * SS * HH * HDIM;   // 4,194,304
        rope_pack<<<total / 256, 256, 0, stream>>>(qf, kf, vf, qpk, kpk, vpk);
    }

    // 4) Flash attention (writes f16 attn in (B,S,D) layout, reusing qf space)
    {
        dim3 grid(SS / 64, HH, BB);
        flash_attn_wmma<<<grid, 128, 0, stream>>>(qpk, kpk, vpk, at16);
    }

    // 5) Output projection: out = attn @ wo^T (fp32 result)
    {
        dim3 grid(DD / 64, MM / 128);
        gemm_awt_wmma<<<grid, 256, 0, stream>>>(at16, wo16, out, MM, DD, DD);
    }
}
